// Net_66907000537771
// MI455X (gfx1250) — matrix operand order repaired and checked by reading, not compiled, not run on hardware
//
#include <hip/hip_runtime.h>
#include <hip/hip_bf16.h>

#define N_NODES 50000
#define E_EDGES 800000

typedef unsigned int  u32;
typedef unsigned short u16;

typedef __attribute__((ext_vector_type(16))) __bf16 v16bf;
typedef __attribute__((ext_vector_type(8)))  float  v8f;

// ---------- helpers ----------
__device__ __forceinline__ u16 f2bf(float f) {
    u32 u = __float_as_uint(f);
    u32 r = 0x7FFFu + ((u >> 16) & 1u);   // round-to-nearest-even
    return (u16)((u + r) >> 16);
}
// order-preserving float<->uint encoding for atomic max
__device__ __forceinline__ u32 fenc(float f) {
    u32 u = __float_as_uint(f);
    return (u & 0x80000000u) ? ~u : (u | 0x80000000u);
}
__device__ __forceinline__ float fdec(u32 u) {
    u = (u & 0x80000000u) ? (u & 0x7FFFFFFFu) : ~u;
    return __uint_as_float(u);
}

// ---------- utility kernels ----------
__global__ void fill_u32_kernel(u32* __restrict__ p, u32 v, int n) {
    int i = blockIdx.x * blockDim.x + threadIdx.x;
    if (i < n) p[i] = v;
}

__global__ void to_bf16_kernel(const float* __restrict__ x, u16* __restrict__ o, int n) {
    int i = blockIdx.x * blockDim.x + threadIdx.x;
    if (i < n) o[i] = f2bf(x[i]);
}

// Pack row-major f32 weights [K x M] into WMMA-B-fragment-contiguous bf16,
// zero-padded to Mpad = NT*16 columns.
// Layout: Bp[ ((kt*NT + nt)*32 + lane)*16 + i ] = bf16( W[kt*32 + g*16 + i][nt*16 + col] )
// with col = lane&15, g = lane>>4  (contiguous k = 16g + i, same k order as the A fragment in gemm_kernel).
__global__ void pack_w_kernel(const float* __restrict__ W, u16* __restrict__ Bp,
                              int K, int M, int NT) {
    int idx = blockIdx.x * blockDim.x + threadIdx.x;
    int total = (K / 32) * NT * 512;
    if (idx >= total) return;
    int i    = idx & 15;
    int lane = (idx >> 4) & 31;
    int t    = idx >> 9;
    int nt   = t % NT;
    int kt   = t / NT;
    int col  = lane & 15;
    int g    = lane >> 4;
    int k = kt * 32 + g * 16 + i;
    int m = nt * 16 + col;
    float v = (m < M) ? W[k * M + m] : 0.0f;
    Bp[idx] = f2bf(v);
}

// ---------- WMMA GEMM: C = A*B ----------
// A: nrows x 128 bf16 row-major. B packed as above.
// One wave -> 16 rows x NT*16 cols. acc = NT x v8f (64 VGPRs for NT=8) => no spills.
template <int NT>
__global__ __launch_bounds__(256)
void gemm_kernel(const u16* __restrict__ A,
                 const u16* __restrict__ Bp,
                 float* __restrict__ C,
                 int nrows, int ldc) {
    const int tiles = (nrows + 15) >> 4;
    int wave = (int)((blockIdx.x * (size_t)blockDim.x + threadIdx.x) >> 5);
    int lane = threadIdx.x & 31;
    if (wave >= tiles) return;          // wave-uniform: EXEC stays all-ones for WMMA
    int row0 = wave << 4;
    int l16 = lane & 15;
    int g   = lane >> 4;

    int arow = row0 + l16;
    if (arow >= nrows) arow = nrows - 1;     // clamp (stores are masked)
    const u16* ap = A + (size_t)arow * 128;

    v8f acc[NT] = {};

    for (int kt = 0; kt < 4; ++kt) {         // K = 128 = 4 * 32
        // A fragment: elems 0..15 = A[row][kt*32 + g*16 ..]
        union { uint4 q[2]; v16bf v; } af;
        af.q[0] = *(const uint4*)(ap + kt * 32 + g * 16);
        af.q[1] = *(const uint4*)(ap + kt * 32 + g * 16 + 8);
#pragma unroll
        for (int nt = 0; nt < NT; ++nt) {
            const uint4* p = (const uint4*)(Bp + (size_t)(((kt * NT + nt) * 32 + lane) * 16));
            union { uint4 q[2]; v16bf v; } bf;
            bf.q[0] = p[0]; bf.q[1] = p[1];
            acc[nt] = __builtin_amdgcn_wmma_f32_16x16x32_bf16(
                false, af.v, false, bf.v, (short)0, acc[nt], false, false);
        }
    }
    // D layout: VGPR r -> row = g*8 + r, col = lane&15
#pragma unroll
    for (int nt = 0; nt < NT; ++nt) {
#pragma unroll
        for (int vr = 0; vr < 8; ++vr) {
            int row = row0 + g * 8 + vr;
            if (row < nrows)
                C[(size_t)row * ldc + nt * 16 + l16] = acc[nt][vr];
        }
    }
}

// ---------- edge kernels (one wave32 per edge) ----------
__global__ void edge_score_kernel(const float* __restrict__ gl,
                                  const float* __restrict__ gr,
                                  const float* __restrict__ att,
                                  const int* __restrict__ src,
                                  const int* __restrict__ dst,
                                  u32* __restrict__ mmax,
                                  float* __restrict__ score,
                                  int E, int H, int F, int ld) {
    int e = (int)((blockIdx.x * (size_t)blockDim.x + threadIdx.x) >> 5);
    int lane = threadIdx.x & 31;
    if (e >= E) return;
    int s = src[e], d = dst[e];
    for (int h = 0; h < H; ++h) {
        float part = 0.0f;
        for (int f = lane; f < F; f += 32) {
            float v = gl[(size_t)s * ld + h * F + f] + gr[(size_t)d * ld + h * F + f];
            v = v > 0.0f ? v : 0.2f * v;               // LeakyReLU
            part += v * att[h * F + f];
        }
        for (int m = 16; m > 0; m >>= 1) part += __shfl_xor(part, m, 32);
        if (lane == 0) {
            score[(size_t)e * H + h] = part;
            atomicMax(&mmax[(size_t)d * H + h], fenc(part));
        }
    }
}

__global__ void edge_exp_kernel(const int* __restrict__ dst,
                                const u32* __restrict__ mmax,
                                float* __restrict__ score,
                                float* __restrict__ denom,
                                int E, int H) {
    int i = blockIdx.x * blockDim.x + threadIdx.x;
    if (i >= E * H) return;
    int e = i / H;
    int h = i - e * H;
    int d = dst[e];
    float m = fdec(mmax[(size_t)d * H + h]);
    float a = expf(score[i] - m);
    score[i] = a;
    atomicAdd(&denom[(size_t)d * H + h], a);
}

__global__ void edge_agg_kernel(const float* __restrict__ gl,
                                const float* __restrict__ score,
                                const int* __restrict__ src,
                                const int* __restrict__ dst,
                                float* __restrict__ agg,
                                int E, int H, int F, int ld) {
    int e = (int)((blockIdx.x * (size_t)blockDim.x + threadIdx.x) >> 5);
    int lane = threadIdx.x & 31;
    if (e >= E) return;
    int s = src[e], d = dst[e];
    for (int h = 0; h < H; ++h) {
        float a = score[(size_t)e * H + h];
        for (int f = lane; f < F; f += 32)
            atomicAdd(&agg[(size_t)d * ld + h * F + f],
                      a * gl[(size_t)s * ld + h * F + f]);
    }
}

// normalize by segment sum + activation; write bf16 for next layer's GEMM and/or f32
// act: 0 = ELU, 1 = ReLU, 2 = none
__global__ void normalize_act_kernel(const float* __restrict__ agg,
                                     const float* __restrict__ denom,
                                     u16* __restrict__ hb_out,
                                     float* __restrict__ f32_out,
                                     int n, int H, int F, int ld, int act) {
    int i = blockIdx.x * blockDim.x + threadIdx.x;
    int hf = H * F;
    if (i >= n * hf) return;
    int node = i / hf;
    int rem  = i - node * hf;
    int h    = rem / F;
    float v = agg[(size_t)node * ld + rem] / (denom[(size_t)node * H + h] + 1e-16f);
    if (act == 0)      v = v > 0.0f ? v : (expf(v) - 1.0f);
    else if (act == 1) v = v > 0.0f ? v : 0.0f;
    if (hb_out)  hb_out[(size_t)node * 128 + rem] = f2bf(v);
    if (f32_out) f32_out[i] = v;
}

// one wave per node; C <= 64
__global__ void log_softmax_kernel(const float* __restrict__ h,
                                   float* __restrict__ out, int n, int C) {
    int node = (int)((blockIdx.x * (size_t)blockDim.x + threadIdx.x) >> 5);
    int lane = threadIdx.x & 31;
    if (node >= n) return;
    int c1 = lane + 32;
    float v0 = (lane < C) ? h[(size_t)node * C + lane] : -3.4e38f;
    float v1 = (c1 < C)   ? h[(size_t)node * C + c1]   : -3.4e38f;
    float m = fmaxf(v0, v1);
    for (int s = 16; s > 0; s >>= 1) m = fmaxf(m, __shfl_xor(m, s, 32));
    float e0 = (lane < C) ? expf(v0 - m) : 0.0f;
    float e1 = (c1 < C)   ? expf(v1 - m) : 0.0f;
    float sum = e0 + e1;
    for (int s = 16; s > 0; s >>= 1) sum += __shfl_xor(sum, s, 32);
    float ls = logf(sum);
    if (lane < C) out[(size_t)node * C + lane] = v0 - m - ls;
    if (c1 < C)   out[(size_t)node * C + c1]   = v1 - m - ls;
}

// ---------- host orchestration ----------
extern "C" void kernel_launch(void* const* d_in, const int* in_sizes, int n_in,
                              void* d_out, int out_size, void* d_ws, size_t ws_size,
                              hipStream_t stream) {
    (void)in_sizes; (void)n_in; (void)out_size; (void)ws_size;
    const float* x    = (const float*)d_in[0];
    const float* Wl1  = (const float*)d_in[1];
    const float* Wr1  = (const float*)d_in[2];
    const float* att1 = (const float*)d_in[3];
    const float* Wl2  = (const float*)d_in[4];
    const float* Wr2  = (const float*)d_in[5];
    const float* att2 = (const float*)d_in[6];
    const float* Wl3  = (const float*)d_in[7];
    const float* Wr3  = (const float*)d_in[8];
    const float* att3 = (const float*)d_in[9];
    const int*   ei   = (const int*)d_in[10];
    const int* src = ei;
    const int* dst = ei + E_EDGES;

    char* ws = (char*)d_ws;
    size_t off = 0;
    auto alloc = [&](size_t bytes) -> void* {
        void* p = ws + off;
        off += (bytes + 255) & ~(size_t)255;
        return p;
    };
    u16*   hb   = (u16*)  alloc((size_t)N_NODES * 128 * 2);
    float* gl   = (float*)alloc((size_t)N_NODES * 128 * 4);
    float* gr   = (float*)alloc((size_t)N_NODES * 128 * 4);
    float* agg  = (float*)alloc((size_t)N_NODES * 128 * 4);
    float* sc   = (float*)alloc((size_t)E_EDGES * 4 * 4);
    u32*   mmax = (u32*)  alloc((size_t)N_NODES * 4 * 4);
    float* den  = (float*)alloc((size_t)N_NODES * 4 * 4);
    u16*   Bp1  = (u16*)  alloc(128 * 128 * 2);
    u16*   Bp2  = (u16*)  alloc(128 * 128 * 2);
    float* h3   = (float*)alloc((size_t)N_NODES * 40 * 4);

    const int BT = 256;

    // input features -> bf16
    {
        int n = N_NODES * 128;
        to_bf16_kernel<<<(n + BT - 1) / BT, BT, 0, stream>>>(x, hb, n);
    }

    auto run_layer = [&](const float* Wl, const float* Wr, const float* att,
                         int H, int F, int Mpad, int act, bool last) {
        int NT = Mpad / 16;
        int ld = Mpad;
        int M  = H * F;
        // pack weights -> fragment layout (zero-padded)
        {
            int n = (128 / 32) * NT * 512;
            pack_w_kernel<<<(n + BT - 1) / BT, BT, 0, stream>>>(Wl, Bp1, 128, M, NT);
            pack_w_kernel<<<(n + BT - 1) / BT, BT, 0, stream>>>(Wr, Bp2, 128, M, NT);
        }
        // init accumulators
        {
            int n = N_NODES * ld;
            fill_u32_kernel<<<(n + BT - 1) / BT, BT, 0, stream>>>((u32*)agg, 0u, n);
            int n2 = N_NODES * H;
            fill_u32_kernel<<<(n2 + BT - 1) / BT, BT, 0, stream>>>((u32*)den, 0u, n2);
            fill_u32_kernel<<<(n2 + BT - 1) / BT, BT, 0, stream>>>(mmax, 0u, n2);
        }
        // WMMA GEMMs: gl = h@Wl, gr = h@Wr
        {
            int waves = (N_NODES + 15) / 16;
            int grid  = (waves * 32 + BT - 1) / BT;
            if (NT == 8) {
                gemm_kernel<8><<<grid, BT, 0, stream>>>(hb, Bp1, gl, N_NODES, ld);
                gemm_kernel<8><<<grid, BT, 0, stream>>>(hb, Bp2, gr, N_NODES, ld);
            } else {
                gemm_kernel<4><<<grid, BT, 0, stream>>>(hb, Bp1, gl, N_NODES, ld);
                gemm_kernel<4><<<grid, BT, 0, stream>>>(hb, Bp2, gr, N_NODES, ld);
            }
        }
        // edge phase
        int egrid = (int)(((size_t)E_EDGES * 32 + BT - 1) / BT);
        edge_score_kernel<<<egrid, BT, 0, stream>>>(gl, gr, att, src, dst, mmax, sc,
                                                    E_EDGES, H, F, ld);
        {
            int n = E_EDGES * H;
            edge_exp_kernel<<<(n + BT - 1) / BT, BT, 0, stream>>>(dst, mmax, sc, den,
                                                                  E_EDGES, H);
        }
        edge_agg_kernel<<<egrid, BT, 0, stream>>>(gl, sc, src, dst, agg,
                                                  E_EDGES, H, F, ld);
        // normalize + activation
        {
            int n = N_NODES * H * F;
            normalize_act_kernel<<<(n + BT - 1) / BT, BT, 0, stream>>>(
                agg, den, last ? nullptr : hb, last ? h3 : nullptr,
                N_NODES, H, F, ld, act);
        }
    };

    run_layer(Wl1, Wr1, att1, 4, 32, 128, /*ELU*/0, false);
    run_layer(Wl2, Wr2, att2, 4, 32, 128, /*ReLU*/1, false);
    run_layer(Wl3, Wr3, att3, 1, 40, 64,  /*none*/2, true);

    {
        int grid = (int)(((size_t)N_NODES * 32 + BT - 1) / BT);
        log_softmax_kernel<<<grid, BT, 0, stream>>>(h3, (float*)d_out, N_NODES, 40);
    }
}
